// MyConv2D_31250182046155
// MI455X (gfx1250) — compile-verified, matrix operand order checked by reading, not run on hardware
//
#include <hip/hip_runtime.h>
#include <hip/hip_bf16.h>

typedef __attribute__((ext_vector_type(16))) _Float16 v16h;
typedef __attribute__((ext_vector_type(8)))  _Float16 v8h;
typedef __attribute__((ext_vector_type(4)))  _Float16 v4h;
typedef __attribute__((ext_vector_type(8)))  float    v8f;
typedef __attribute__((ext_vector_type(4)))  float    v4f;

#define H_DIM 112
#define W_DIM 112
#define CIN   64
#define COUT  128
#define ROWS  4                       // output rows per block
#define SROWS (ROWS + 2)              // staged input rows (with halo)
#define SLAB_ELEMS (SROWS * 18 * 64)  // f16 elements in LDS slab
// packed weights: 9 taps x 2 cin-chunks x 8 n-tiles x 32 lanes x 16 halfs
#define WPACK_ELEMS (9 * 2 * 8 * 32 * 16)

// ---------------------------------------------------------------------------
// Repack fp32 HWIO weights into f16 WMMA B-fragment order.
// Linear index: ((((rs*2 + cc)*8 + nt)*32 + lane)*16 + e)
//   k (within 32-chunk) = 8*(lane>>4) + (e&7) + 16*(e>>3), cin = 32*cc + k
//   n = nt*16 + (lane&15)
// ---------------------------------------------------------------------------
__global__ __launch_bounds__(256) void conv_repack_w(
    const float* __restrict__ kern, _Float16* __restrict__ wpk) {
  int idx = blockIdx.x * 256 + threadIdx.x;
  if (idx >= WPACK_ELEMS) return;
  int e    = idx & 15;
  int lane = (idx >> 4) & 31;
  int nt   = (idx >> 9) & 7;
  int cc   = (idx >> 12) & 1;
  int rs   = idx >> 13;                       // kh*3 + kw
  int kk   = 32 * cc + 8 * (lane >> 4) + (e & 7) + 16 * (e >> 3);
  int n    = nt * 16 + (lane & 15);
  wpk[idx] = (_Float16)kern[(rs * CIN + kk) * COUT + n];
}

// ---------------------------------------------------------------------------
// Implicit-GEMM conv: block = (16-wide w tile, 4 h rows, one image).
// 8 waves; wave i handles Cout tile i (16 couts) for all 4 row-tiles.
// 72 WMMAs per wave; B fragments register-reused across the 4 M tiles.
// ---------------------------------------------------------------------------
__global__ __launch_bounds__(256) void conv_wmma_kernel(
    const float* __restrict__ x, const _Float16* __restrict__ wpk,
    const float* __restrict__ bias, float* __restrict__ out) {
  __shared__ __align__(16) _Float16 slab[SLAB_ELEMS];

  const int w0  = blockIdx.x * 16;
  const int h0  = blockIdx.y * ROWS;
  const int b   = blockIdx.z;
  const int tid = threadIdx.x;

  // ---- stage padded input slab (f32 -> f16), 4 channels per iteration ----
  for (int t = tid; t < SLAB_ELEMS / 4; t += 256) {
    int c4  = t & 15;                  // channel group of 4
    int wpi = (t >> 4) % 18;           // padded w index 0..17
    int r   = t / (18 * 16);           // padded row 0..SROWS-1
    int hh  = h0 + r - 1;
    int ww  = w0 + wpi - 1;
    v4f v = {0.0f, 0.0f, 0.0f, 0.0f};
    if (hh >= 0 && hh < H_DIM && ww >= 0 && ww < W_DIM)
      v = *(const v4f*)&x[(((size_t)b * H_DIM + hh) * W_DIM + ww) * CIN + c4 * 4];
    v4h hv;
    hv.x = (_Float16)v.x; hv.y = (_Float16)v.y;
    hv.z = (_Float16)v.z; hv.w = (_Float16)v.w;
    *(v4h*)&slab[(r * 18 + wpi) * 64 + c4 * 4] = hv;
  }
  __syncthreads();

  const int wave = tid >> 5;           // Cout tile 0..7
  const int lane = tid & 31;
  const int lm   = lane & 15;          // A: pixel m; B/C: column n
  const int lh   = lane >> 4;          // half-wave selector

  v8f acc[ROWS] = {};

#pragma unroll
  for (int rs = 0; rs < 9; ++rs) {
    const int kh = rs / 3;
    const int kw = rs % 3;
#pragma unroll
    for (int cc = 0; cc < 2; ++cc) {
      // B fragment: one 32B contiguous read per lane, reused for all 4 M tiles
      v16h bf = *(const v16h*)&wpk[((((rs * 2 + cc) * 8 + wave) * 32) + lane) * 16];
#pragma unroll
      for (int mt = 0; mt < ROWS; ++mt) {
        // A base: pixel (w0+lm), output row h0+mt, tap (kh,kw)
        const _Float16* abase =
            &slab[((kh + mt) * 18 + lm + kw) * 64 + 32 * cc + 8 * lh];
        v8h  alo = *(const v8h*)(abase);        // K = 8*lh + 0..7
        v8h  ahi = *(const v8h*)(abase + 16);   // K = 16 + 8*lh + 0..7
        v16h a   = __builtin_shufflevector(alo, ahi, 0, 1, 2, 3, 4, 5, 6, 7,
                                           8, 9, 10, 11, 12, 13, 14, 15);
        acc[mt] = __builtin_amdgcn_wmma_f32_16x16x32_f16(
            /*neg_a=*/false, a, /*neg_b=*/false, bf,
            /*c_mod=*/(short)0, acc[mt], /*reuse_a=*/false, /*reuse_b=*/false);
      }
    }
  }

  // ---- epilogue: bias + store (C/D layout: n = lane&15, m = r + 8*lh) ----
  const int   n  = wave * 16 + lm;
  const float bv = bias[n];
#pragma unroll
  for (int mt = 0; mt < ROWS; ++mt) {
    const size_t ob =
        (((size_t)b * H_DIM + (h0 + mt)) * W_DIM + w0) * COUT + n;
#pragma unroll
    for (int r = 0; r < 8; ++r) {
      out[ob + (size_t)(r + 8 * lh) * COUT] = acc[mt][r] + bv;
    }
  }
}

extern "C" void kernel_launch(void* const* d_in, const int* in_sizes, int n_in,
                              void* d_out, int out_size, void* d_ws,
                              size_t ws_size, hipStream_t stream) {
  const float* x    = (const float*)d_in[0];
  const float* kern = (const float*)d_in[1];
  const float* bias = (const float*)d_in[2];
  float*       out  = (float*)d_out;
  _Float16*    wpk  = (_Float16*)d_ws;   // 147456 bytes of scratch

  conv_repack_w<<<(WPACK_ELEMS + 255) / 256, 256, 0, stream>>>(kern, wpk);

  dim3 grid(W_DIM / 16, H_DIM / ROWS, 32);
  conv_wmma_kernel<<<grid, 256, 0, stream>>>(x, wpk, bias, out);
}
